// SNN_22101901705844
// MI455X (gfx1250) — matrix operand order repaired and checked by reading, not compiled, not run on hardware
//
#include <hip/hip_runtime.h>
#include <hip/hip_bf16.h>
#include <math.h>

#define B_ 128
#define I_ 1024
#define H_ 4096
#define O_ 512
#define T_ 100

// SNN params (hid == out): vdecay, vth, grad_win
#define VDECAY 0.5f
#define VTH    0.3f
#define GWIN   0.3f

typedef __bf16 bf16_t;
typedef __attribute__((ext_vector_type(16))) __bf16 v16bf;
typedef __attribute__((ext_vector_type(8)))  float  v8f;
typedef __attribute__((ext_vector_type(4)))  int    vi4;

union FragBF { v16bf v; uint4 u[2]; };
union Pack8  { uint4 u; bf16_t h[8]; };
union F16x   { float4 v[4]; float f[16]; };

#define AS1 __attribute__((address_space(1)))
#define AS3 __attribute__((address_space(3)))

// CDNA5 async global->LDS path (ASYNCcnt-tracked), with safe fallback.
#if defined(__has_builtin)
#if __has_builtin(__builtin_amdgcn_global_load_async_to_lds_b128)
#define SNN_ASYNC 1
#endif
#endif

#if defined(SNN_ASYNC)
__device__ __forceinline__ void async_b128(const void* g, void* l) {
    __builtin_amdgcn_global_load_async_to_lds_b128(
        (AS1 vi4*)g, (AS3 vi4*)l, 0, 0);
}
__device__ __forceinline__ void wait_async0() {
#if __has_builtin(__builtin_amdgcn_s_wait_asynccnt)
    __builtin_amdgcn_s_wait_asynccnt(0);
#else
    asm volatile("s_wait_asynccnt 0" ::: "memory");
#endif
}
#endif

// float atomic max via signed/unsigned int atomics (target init'd to -inf bits)
__device__ __forceinline__ void atomicMaxF(float* addr, float val) {
    if (val >= 0.0f) atomicMax((int*)addr, __float_as_int(val));
    else             atomicMin((unsigned int*)addr, (unsigned int)__float_as_int(val));
}

// ---------------------------------------------------------------------------
// Fused LDS-double-buffered WMMA GEMM step.
//   state = (A@Bw^T) + state*VDECAY*(1-spike_prev),
//   spike_prev = state > VTH*tanh(GWIN * *prev_max)
//   *cur_max = max(state_new) (atomic); HAS_SUM: sum_acc += spike_prev.
// A_SPIKE: A is f32 voltage, binarized on the fly against VTH*tanh(GWIN * *a_max).
// Block = 128 threads (4 waves), block tile 64(M)x64(N); wave = 16x64 strip.
// lda == K for both callers.
// ---------------------------------------------------------------------------
template <int N, int K, bool A_SPIKE, bool HAS_SUM>
__global__ void __launch_bounds__(128)
snn_gemm_step(const void* __restrict__ Aptr,
              const bf16_t* __restrict__ Bw,
              const float* __restrict__ a_max,
              float* __restrict__ state,
              const float* __restrict__ prev_max,
              float* __restrict__ cur_max,
              float* __restrict__ sum_acc)
{
    constexpr int LROW = 40;        // 32 K halfwords + 8 pad = 80B rows (16B aligned)
    constexpr int C = K >> 5;       // number of 32-K chunks
    constexpr int nBlocks = N >> 6;

    __shared__ __attribute__((aligned(16))) bf16_t lA[2][64 * LROW];
    __shared__ __attribute__((aligned(16))) bf16_t lB[2][64 * LROW];

    const int tid = threadIdx.x, lane = tid & 31, w = tid >> 5;
    const int m0g = (blockIdx.x / nBlocks) * 64;
    const int n0g = (blockIdx.x % nBlocks) * 64;

    const float thr_prev = VTH * tanhf(GWIN * prev_max[0]);
    float thr_a = 0.0f;
    if constexpr (A_SPIKE) thr_a = VTH * tanhf(GWIN * a_max[0]);
    const bf16_t ONE = (bf16_t)1.0f, ZERO = (bf16_t)0.0f;

    // staging: each thread moves one 16-K-value piece (32B bf16) of A and of B
    const int    srow  = tid >> 1;
    const int    shalf = (tid & 1) * 16;
    const size_t a_off = (size_t)(m0g + srow) * K + shalf;
    const size_t b_off = (size_t)(n0g + srow) * K + shalf;
    const int    l_off = srow * LROW + shalf;

    uint4 ra0, ra1, rb0, rb1;          // register-staging storage
    (void)rb0; (void)rb1;

    auto begin_stage = [&](int buf, int kb) {
#if defined(SNN_ASYNC)
        async_b128(Bw + b_off + kb,     &lB[buf][l_off]);
        async_b128(Bw + b_off + kb + 8, &lB[buf][l_off + 8]);
        if constexpr (!A_SPIKE) {
            const bf16_t* p = (const bf16_t*)Aptr + a_off + kb;
            async_b128(p,     &lA[buf][l_off]);
            async_b128(p + 8, &lA[buf][l_off + 8]);
        }
#else
        {
            const bf16_t* q = Bw + b_off + kb;
            rb0 = ((const uint4*)q)[0];
            rb1 = ((const uint4*)q)[1];
        }
        if constexpr (!A_SPIKE) {
            const bf16_t* p = (const bf16_t*)Aptr + a_off + kb;
            ra0 = ((const uint4*)p)[0];
            ra1 = ((const uint4*)p)[1];
        }
#endif
        if constexpr (A_SPIKE) {      // f32 voltage -> 0/1 bf16, via registers
            const float* p = (const float*)Aptr + a_off + kb;
            F16x x;
            x.v[0] = ((const float4*)p)[0];
            x.v[1] = ((const float4*)p)[1];
            x.v[2] = ((const float4*)p)[2];
            x.v[3] = ((const float4*)p)[3];
            Pack8 q0, q1;
#pragma unroll
            for (int j = 0; j < 8; ++j) {
                q0.h[j] = (x.f[j]     > thr_a) ? ONE : ZERO;
                q1.h[j] = (x.f[j + 8] > thr_a) ? ONE : ZERO;
            }
            ra0 = q0.u; ra1 = q1.u;
        }
    };
    auto end_stage = [&](int buf) {
        if constexpr (A_SPIKE) {
            ((uint4*)&lA[buf][l_off])[0] = ra0;
            ((uint4*)&lA[buf][l_off])[1] = ra1;
        }
#if defined(SNN_ASYNC)
        wait_async0();
#else
        if constexpr (!A_SPIKE) {
            ((uint4*)&lA[buf][l_off])[0] = ra0;
            ((uint4*)&lA[buf][l_off])[1] = ra1;
        }
        ((uint4*)&lB[buf][l_off])[0] = rb0;
        ((uint4*)&lB[buf][l_off])[1] = rb1;
#endif
    };

    // WMMA fragment LDS offsets (halfwords)
    const int afrow  = (w * 16 + (lane & 15)) * LROW + (lane >> 4) * 8;
    const int bfrow0 = (lane & 15) * LROW + (lane >> 4) * 8;

    v8f acc[4] = {};
    auto compute = [&](int buf) {
        FragBF a, b[4];
        const bf16_t* ap = &lA[buf][afrow];
        a.u[0] = ((const uint4*)ap)[0];
        a.u[1] = *(const uint4*)(ap + 16);
#pragma unroll
        for (int nt = 0; nt < 4; ++nt) {
            const bf16_t* bp = &lB[buf][bfrow0 + nt * 16 * LROW];
            b[nt].u[0] = ((const uint4*)bp)[0];
            b[nt].u[1] = *(const uint4*)(bp + 16);
        }
#pragma unroll
        for (int nt = 0; nt < 4; ++nt)
            acc[nt] = __builtin_amdgcn_wmma_f32_16x16x32_bf16(
                false, a.v, false, b[nt].v, (short)0, acc[nt], false, false);
    };

    begin_stage(0, 0);
    end_stage(0);
    __syncthreads();

#pragma unroll 2
    for (int c = 0; c < C - 1; ++c) {
        const int cur = c & 1;
        begin_stage(cur ^ 1, (c + 1) << 5);
        {   // branch-free clamped prefetch, near-cache temporal
            const int kb2 = ((c + 2 < C) ? (c + 2) : (C - 1)) << 5;
            if constexpr (A_SPIKE) __builtin_prefetch((const float*)Aptr + a_off + kb2, 0, 3);
            else                   __builtin_prefetch((const bf16_t*)Aptr + a_off + kb2, 0, 3);
            __builtin_prefetch(Bw + b_off + kb2, 0, 3);
        }
        compute(cur);
        end_stage(cur ^ 1);
        __syncthreads();
    }
    compute((C - 1) & 1);

    // Epilogue: leaky-integrate with previous-spike reset; track max; accumulate spikes
    float lmax = -__builtin_inff();
    const int rbase = m0g + w * 16 + ((lane >> 4) << 3);
    const int cbase = n0g + (lane & 15);
#pragma unroll
    for (int nt = 0; nt < 4; ++nt) {
#pragma unroll
        for (int r = 0; r < 8; ++r) {
            const size_t idx = (size_t)(rbase + r) * (size_t)N + (size_t)(cbase + nt * 16);
            const float vprev = state[idx];
            const float sprev = (vprev > thr_prev) ? 1.0f : 0.0f;
            const float vnew  = acc[nt][r] + vprev * (VDECAY * (1.0f - sprev));
            state[idx] = vnew;
            lmax = fmaxf(lmax, vnew);
            if constexpr (HAS_SUM) sum_acc[idx] += sprev;
        }
    }
#pragma unroll
    for (int off = 16; off >= 1; off >>= 1)
        lmax = fmaxf(lmax, __shfl_xor(lmax, off, 32));
    if (lane == 0) atomicMaxF(cur_max, lmax);
}

// spike_data [B, I, T] f32  ->  xbf [T, B, I] bf16 (K-contiguous for WMMA A loads)
__global__ void transpose_cvt_kernel(const float* __restrict__ in,
                                     bf16_t* __restrict__ out)
{
    const int idx = blockIdx.x * blockDim.x + threadIdx.x;
    if (idx >= T_ * B_ * I_) return;
    const int i = idx % I_;
    const int b = (idx / I_) % B_;
    const int t = idx / (I_ * B_);
    out[idx] = (bf16_t)in[((size_t)b * I_ + i) * T_ + t];
}

__global__ void cvt_bf16_kernel(const float* __restrict__ in,
                                bf16_t* __restrict__ out, int n)
{
    const int i = blockIdx.x * blockDim.x + threadIdx.x;
    if (i < n) out[i] = (bf16_t)in[i];
}

// Init voltage state from inputs, zero output sum, set max sentinels.
__global__ void init_kernel(float* __restrict__ hv, const float* __restrict__ hv0,
                            float* __restrict__ ov, const float* __restrict__ ov0,
                            float* __restrict__ outsum,
                            float* __restrict__ hmax, float* __restrict__ omax)
{
    const int i = blockIdx.x * blockDim.x + threadIdx.x;
    if (i < B_ * H_) hv[i] = hv0[i];
    if (i < B_ * O_) { ov[i] = ov0[i]; outsum[i] = 0.0f; }
    if (i <= T_) {
        // index 0 = "step -1" sentinel: volt==0 and thr==0 -> spike==0 (matches *_spike0)
        const float v = (i == 0) ? 0.0f : __int_as_float((int)0xFF800000u);
        hmax[i] = v;
        omax[i] = v;
    }
}

// Add the final timestep's output spikes into the sum.
__global__ void final_spike_kernel(const float* __restrict__ ov,
                                   const float* __restrict__ maxp,
                                   float* __restrict__ outsum)
{
    const float thr = VTH * tanhf(GWIN * maxp[0]);
    const int i = blockIdx.x * blockDim.x + threadIdx.x;
    if (i < B_ * O_) outsum[i] += (ov[i] > thr) ? 1.0f : 0.0f;
}

extern "C" void kernel_launch(void* const* d_in, const int* in_sizes, int n_in,
                              void* d_out, int out_size, void* d_ws, size_t ws_size,
                              hipStream_t stream)
{
    (void)in_sizes; (void)n_in; (void)out_size; (void)ws_size;

    const float* spike_data = (const float*)d_in[0];   // [B, I, T]
    const float* hid_volt0  = (const float*)d_in[2];   // [B, H]
    const float* out_volt0  = (const float*)d_in[4];   // [B, O]
    const float* w_hid      = (const float*)d_in[5];   // [H, I]
    const float* w_out      = (const float*)d_in[6];   // [O, H]
    float* outsum = (float*)d_out;                     // [B, O]

    // --- workspace layout (256B aligned pieces) ---
    char* w = (char*)d_ws;
    float*  hv   = (float*)w;  w += (size_t)B_ * H_ * 4;   // 2 MB
    float*  ov   = (float*)w;  w += (size_t)B_ * O_ * 4;   // 256 KB
    float*  hmax = (float*)w;  w += 512;                   // T_+1 floats
    float*  omax = (float*)w;  w += 512;
    bf16_t* whb  = (bf16_t*)w; w += (size_t)H_ * I_ * 2;   // 8 MB
    bf16_t* wob  = (bf16_t*)w; w += (size_t)O_ * H_ * 2;   // 4 MB
    bf16_t* xbf  = (bf16_t*)w;                             // 26.2 MB [T,B,I]

    // --- one-time prep (graph-capture safe, deterministic) ---
    init_kernel<<<(B_ * H_ + 255) / 256, 256, 0, stream>>>(hv, hid_volt0, ov, out_volt0,
                                                           outsum, hmax, omax);
    cvt_bf16_kernel<<<(H_ * I_ + 255) / 256, 256, 0, stream>>>(w_hid, whb, H_ * I_);
    cvt_bf16_kernel<<<(O_ * H_ + 255) / 256, 256, 0, stream>>>(w_out, wob, O_ * H_);
    transpose_cvt_kernel<<<(T_ * B_ * I_ + 255) / 256, 256, 0, stream>>>(spike_data, xbf);

    // --- recurrence: 2 fused kernels per step ---
    const int k1_blocks = (B_ / 64) * (H_ / 64);   // 128
    const int k2_blocks = (B_ / 64) * (O_ / 64);   // 16

    for (int t = 0; t < T_; t++) {
        const bf16_t* xt = xbf + (size_t)t * B_ * I_;
        // hidden: hv = xt @ w_hid^T + hv*decay*(1-hs_prev); hmax[t+1] = max(hv)
        snn_gemm_step<H_, I_, false, false><<<k1_blocks, 128, 0, stream>>>(
            xt, whb, nullptr, hv, hmax + t, hmax + t + 1, nullptr);
        // output: binarize hv on the fly (thr from hmax[t+1]) as GEMM-A;
        //         ov = hs @ w_out^T + ov*decay*(1-os_prev); omax[t+1] = max(ov);
        //         outsum += os_prev (step t-1 spikes; step -1 spikes are 0)
        snn_gemm_step<O_, H_, true, true><<<k2_blocks, 128, 0, stream>>>(
            hv, wob, hmax + t + 1, ov, omax + t, omax + t + 1, outsum);
    }
    // add final step's output spikes
    final_spike_kernel<<<(B_ * O_ + 255) / 256, 256, 0, stream>>>(ov, omax + T_, outsum);
}
